// VQVAE_12532714570171
// MI455X (gfx1250) — matrix operand order repaired and checked by reading, not compiled, not run on hardware
//
#include <hip/hip_runtime.h>

typedef __attribute__((ext_vector_type(16))) __bf16 v16bf;
typedef __attribute__((ext_vector_type(8)))  float   v8f;

#define NUM_CODES 256
#define DIM       128
#define ROWS_BLK  128           // rows of z per block (8 waves x 16 rows)
#define N_ROWS    (32 * 8192)   // 262144

// ---------------------------------------------------------------------------
// Prep: codebook f32 -> bf16 (workspace) + (-0.5 * ||e||^2) per code
// grid = 256 blocks (one per code), block = 128 threads (one per element)
// ---------------------------------------------------------------------------
__global__ void vq_prep(const float* __restrict__ cb,
                        __bf16* __restrict__ cb_bf,
                        float* __restrict__ nrm_scaled) {
    __shared__ float red[DIM];
    const int code = blockIdx.x;
    const int t    = threadIdx.x;
    float v = cb[code * DIM + t];
    cb_bf[code * DIM + t] = (__bf16)v;
    red[t] = v * v;
    __syncthreads();
    for (int s = DIM / 2; s > 0; s >>= 1) {
        if (t < s) red[t] += red[t + s];
        __syncthreads();
    }
    if (t == 0) nrm_scaled[code] = -0.5f * red[0];
}

// ---------------------------------------------------------------------------
// Main: per-row nearest code via WMMA, gather z_q, indices, partial loss
// grid = N_ROWS/ROWS_BLK = 2048 blocks, block = 256 threads (8 wave32)
// ---------------------------------------------------------------------------
__global__ __launch_bounds__(256)
void vq_main(const float* __restrict__ z,
             const float* __restrict__ cb,          // f32 codebook (exact)
             const __bf16* __restrict__ cb_bf_g,    // bf16 codebook (ws)
             const float* __restrict__ nrm_g,       // -0.5*||e||^2 (ws)
             float* __restrict__ out_zq,
             float* __restrict__ out_idx,
             float* __restrict__ partials) {
    __shared__ __align__(16) float  lds_z[ROWS_BLK * DIM];     // 64 KB
    __shared__ __align__(16) __bf16 lds_cb[NUM_CODES * DIM];   // 64 KB
    __shared__ float lds_nrm[NUM_CODES];
    __shared__ int   lds_idx[ROWS_BLK];
    __shared__ float lds_red[256];

    const int t = threadIdx.x;

    // ---- Phase 0: cooperative staging into LDS ----
    {
        const float4* zg = (const float4*)(z + (size_t)blockIdx.x * ROWS_BLK * DIM);
        float4* zl = (float4*)lds_z;
        #pragma unroll
        for (int i = 0; i < 16; ++i) zl[t + i * 256] = zg[t + i * 256];

        const float4* cg = (const float4*)cb_bf_g;     // 64 KB = 4096 float4
        float4* cl = (float4*)lds_cb;
        #pragma unroll
        for (int i = 0; i < 16; ++i) cl[t + i * 256] = cg[t + i * 256];

        lds_nrm[t] = nrm_g[t];
    }
    __syncthreads();

    // ---- Phase 1: WMMA scores  (score = z.e - 0.5||e||^2) ----
    const int lane = t & 31;
    const int wave = t >> 5;
    const int m    = lane & 15;   // row (A) / column (B,C) within tile
    const int hi   = lane >> 4;   // half-wave selector
    const int rowL = wave * 16 + m;

    v8f acc[16];
    #pragma unroll
    for (int c = 0; c < 16; ++c) {
        float ci = lds_nrm[c * 16 + m];             // per-column C init
        acc[c] = (v8f){ci, ci, ci, ci, ci, ci, ci, ci};
    }

    #pragma unroll
    for (int kk = 0; kk < 4; ++kk) {
        const int k0 = kk * 32;
        // A fragment: lane holds 16 contiguous K at k0+hi*16 (matches B)
        const float* zp = &lds_z[rowL * DIM + k0 + hi * 16];
        v16bf a;
        #pragma unroll
        for (int i = 0; i < 8; ++i) {
            a[i]     = (__bf16)zp[i];
            a[8 + i] = (__bf16)zp[8 + i];
        }
        #pragma unroll
        for (int c = 0; c < 16; ++c) {
            // B fragment: 32x16 layout -> lane holds 16 contiguous K at k0+hi*16
            const v16bf b = *(const v16bf*)&lds_cb[(c * 16 + m) * DIM + k0 + hi * 16];
            acc[c] = __builtin_amdgcn_wmma_f32_16x16x32_bf16(
                false, a, false, b, (short)0, acc[c], false, false);
        }
    }

    // ---- Phase 2: argmax over 256 codes per row ----
    float bv[8]; int bi[8];
    #pragma unroll
    for (int j = 0; j < 8; ++j) { bv[j] = acc[0][j]; bi[j] = m; }
    #pragma unroll
    for (int c = 1; c < 16; ++c) {
        const int id = c * 16 + m;
        #pragma unroll
        for (int j = 0; j < 8; ++j) {
            float v = acc[c][j];
            if (v > bv[j]) { bv[j] = v; bi[j] = id; }   // ties keep lower id
        }
    }
    // cross-lane reduce within each 16-lane half (C layout: lanes 0-15 -> M=j,
    // lanes 16-31 -> M=j+8); xor masks 1..8 never cross the halves.
    for (int off = 1; off < 16; off <<= 1) {
        #pragma unroll
        for (int j = 0; j < 8; ++j) {
            float ov = __shfl_xor(bv[j], off, 32);
            int   oi = __shfl_xor(bi[j], off, 32);
            if (ov > bv[j] || (ov == bv[j] && oi < bi[j])) { bv[j] = ov; bi[j] = oi; }
        }
    }
    if (m == 0) {
        #pragma unroll
        for (int j = 0; j < 8; ++j) lds_idx[wave * 16 + hi * 8 + j] = bi[j];
    }
    __syncthreads();

    // ---- Phase 3: gather exact codebook rows, write outputs, local loss ----
    const int r = t >> 1;           // local row
    const int h = t & 1;            // half of the 128-dim vector
    const int idx = lds_idx[r];
    const size_t grow = (size_t)blockIdx.x * ROWS_BLK + r;

    const float4* e4 = (const float4*)(cb + (size_t)idx * DIM + h * 64);
    const float4* z4 = (const float4*)(&lds_z[r * DIM + h * 64]);
    float4* o4 = (float4*)(out_zq + grow * DIM + h * 64);

    float s = 0.0f;
    #pragma unroll
    for (int i = 0; i < 16; ++i) {
        float4 ev = e4[i];
        float4 zv = z4[i];
        o4[i] = ev;                                  // z_q_st == z_q numerically
        float dx = ev.x - zv.x, dy = ev.y - zv.y;
        float dz = ev.z - zv.z, dw = ev.w - zv.w;
        s += dx * dx + dy * dy + dz * dz + dw * dw;
    }
    if (h == 0) out_idx[grow] = (float)idx;

    // ---- Phase 4: deterministic block reduction of squared error ----
    lds_red[t] = s;
    __syncthreads();
    for (int stp = 128; stp > 0; stp >>= 1) {
        if (t < stp) lds_red[t] += lds_red[t + stp];
        __syncthreads();
    }
    if (t == 0) partials[blockIdx.x] = lds_red[0];
}

// ---------------------------------------------------------------------------
// Finalize: loss = 1.25 * sum(partials) / (N*D)
// ---------------------------------------------------------------------------
__global__ void vq_finalize(const float* __restrict__ partials,
                            float* __restrict__ loss_out,
                            int n, float scale) {
    __shared__ float sm[256];
    float s = 0.0f;
    for (int i = threadIdx.x; i < n; i += 256) s += partials[i];
    sm[threadIdx.x] = s;
    __syncthreads();
    for (int stp = 128; stp > 0; stp >>= 1) {
        if (threadIdx.x < stp) sm[threadIdx.x] += sm[threadIdx.x + stp];
        __syncthreads();
    }
    if (threadIdx.x == 0) loss_out[0] = sm[0] * scale;
}

extern "C" void kernel_launch(void* const* d_in, const int* in_sizes, int n_in,
                              void* d_out, int out_size, void* d_ws, size_t ws_size,
                              hipStream_t stream) {
    const float* z  = (const float*)d_in[0];   // [32*8192, 128] f32
    const float* cb = (const float*)d_in[1];   // [256, 128] f32
    float* out = (float*)d_out;

    char* ws = (char*)d_ws;
    __bf16* cb_bf   = (__bf16*)ws;                       // 64 KB
    float*  nrm     = (float*)(ws + 65536);              // 1 KB
    float*  partials = (float*)(ws + 65536 + 1024);      // 8 KB

    const size_t ND = (size_t)N_ROWS * DIM;              // 33554432
    float* loss_out = out + ND;
    float* idx_out  = out + ND + 1;

    const int nblk = N_ROWS / ROWS_BLK;                  // 2048

    vq_prep<<<NUM_CODES, DIM, 0, stream>>>(cb, cb_bf, nrm);
    vq_main<<<nblk, 256, 0, stream>>>(z, cb, cb_bf, nrm, out, idx_out, partials);
    vq_finalize<<<1, 256, 0, stream>>>(partials, loss_out, nblk,
                                       1.25f / (float)ND);
}
